// CausalMultiHeadSelfAttention_11158325035245
// MI455X (gfx1250) — matrix operand order repaired and checked by reading, not compiled, not run on hardware
//
#include <hip/hip_runtime.h>

// ---------------------------------------------------------------------------
// CDNA5 (gfx1250) causal MHA forward: bf16 WMMA pipeline.
// qkv-GEMM -> RoPE -> flash attention (async LDS tiles + v_wmma) -> out-GEMM.
// ---------------------------------------------------------------------------

typedef __attribute__((ext_vector_type(16))) __bf16 v16bf;
typedef __attribute__((ext_vector_type(8)))  float  v8f;

struct __attribute__((aligned(16))) U4 { unsigned x, y, z, w; };
union FragBF { U4 u[2]; v16bf v; };

__device__ __forceinline__ unsigned short f2bf_u(float f) {
  union { float f; unsigned u; } x; x.f = f;
  unsigned r = x.u + 0x7FFFu + ((x.u >> 16) & 1u);   // round-to-nearest-even
  return (unsigned short)(r >> 16);
}
__device__ __forceinline__ float bf2f(unsigned short s) {
  union { unsigned u; float f; } x; x.u = ((unsigned)s) << 16;
  return x.f;
}

static constexpr int Bb = 4, Ss = 2048, Dd = 1024, Hh = 16, DH = 64;
static constexpr int MB = Bb * Ss;          // 8192 rows
static constexpr float NEG_BIG = -1.0e30f;

__device__ __forceinline__ v8f wmma_bf16(const FragBF& a, const FragBF& b, v8f c) {
  return __builtin_amdgcn_wmma_f32_16x16x32_bf16(
      /*neg_a=*/false, a.v, /*neg_b=*/false, b.v,
      /*c_mod=*/(short)0, c, /*reuse_a=*/false, /*reuse_b=*/false);
}

// --- CDNA5 async global->LDS copy (ASYNCcnt-tracked, no VGPR staging) ------
typedef __attribute__((address_space(3))) char lds_char;

__device__ __forceinline__ unsigned lds_offset(void* p) {
  return (unsigned)(unsigned long long)(lds_char*)p;
}
__device__ __forceinline__ void async_copy_b128(void* lds_dst, const void* gsrc) {
  asm volatile("global_load_async_to_lds_b128 %0, %1, off"
               :
               : "v"(lds_offset(lds_dst)), "v"((unsigned long long)gsrc)
               : "memory");
}
__device__ __forceinline__ void wait_async0() {
#if __has_builtin(__builtin_amdgcn_s_wait_asynccnt)
  __builtin_amdgcn_s_wait_asynccnt(0);
#else
  asm volatile("s_wait_asynccnt 0" ::: "memory");
#endif
}

// ---------------------------------------------------------------------------
// 1) fp32 -> bf16 conversion
// ---------------------------------------------------------------------------
__global__ void convert_f32_bf16(const float* __restrict__ in,
                                 unsigned short* __restrict__ out, int n) {
  int i = blockIdx.x * blockDim.x + threadIdx.x;
  if (i < n) out[i] = f2bf_u(in[i]);
}

// ---------------------------------------------------------------------------
// 2) QKV projection: C[m][n] = sum_k X[m][k] * W[n][k]
//    M=8192, N=3072 (=3*D, n = kqv*D + o), K=1024. Stored as qkv[kqv][m][o].
//    Wave tile: 32 x 64 (2 A-frags, shared B-frags, 8 WMMA / k-step).
// ---------------------------------------------------------------------------
__global__ void gemm_qkv(const unsigned short* __restrict__ xb,
                         const unsigned short* __restrict__ wb,
                         unsigned short* __restrict__ qkvb) {
  const int w    = threadIdx.x >> 5;
  const int lane = threadIdx.x & 31;
  const int lr   = lane & 15;
  const int hi   = lane >> 4;
  const int m0 = (blockIdx.x * 4 + w) * 32;
  const int n0 = blockIdx.y * 64;

  const v8f z8 = {0.f,0.f,0.f,0.f,0.f,0.f,0.f,0.f};
  v8f acc[2][4];
#pragma unroll
  for (int r = 0; r < 2; ++r)
#pragma unroll
    for (int t = 0; t < 4; ++t) acc[r][t] = z8;

  for (int k = 0; k < Dd; k += 32) {
    FragBF a[2];
#pragma unroll
    for (int r = 0; r < 2; ++r) {
      const unsigned short* ap = xb + (size_t)(m0 + r * 16 + lr) * Dd + k + hi * 8;
      a[r].u[0] = *(const U4*)ap;
      a[r].u[1] = *(const U4*)(ap + 16);
    }
#pragma unroll
    for (int t = 0; t < 4; ++t) {
      FragBF bf;
      const unsigned short* bp = wb + (size_t)(n0 + t * 16 + lr) * Dd + k + hi * 8;
      bf.u[0] = *(const U4*)bp;
      bf.u[1] = *(const U4*)(bp + 16);
      acc[0][t] = wmma_bf16(a[0], bf, acc[0][t]);
      acc[1][t] = wmma_bf16(a[1], bf, acc[1][t]);
    }
  }

  const int kqv  = n0 >> 10;          // which of q/k/v
  const int noff = n0 & 1023;         // column within D
  unsigned short* out = qkvb + (size_t)kqv * MB * Dd;
#pragma unroll
  for (int r = 0; r < 2; ++r)
#pragma unroll
    for (int t = 0; t < 4; ++t)
#pragma unroll
      for (int j = 0; j < 8; ++j) {
        int row = m0 + r * 16 + j + 8 * hi;
        int col = noff + t * 16 + lr;
        out[(size_t)row * Dd + col] = f2bf_u(acc[r][t][j]);
      }
}

// ---------------------------------------------------------------------------
// 3) RoPE on q and k (they sit contiguously: rows [0, 2*MB) of qkv buffer)
// ---------------------------------------------------------------------------
__global__ void rope_kernel(unsigned short* qk, const int* __restrict__ pos) {
  int idx = blockIdx.x * blockDim.x + threadIdx.x;  // one thread per pair
  int m  = idx >> 9;                                // row in [0, 2*MB)
  int dp = idx & 511;                               // pair index within D
  int s  = m & (Ss - 1);                            // sequence position
  int i  = dp & 31;                                 // pair index within head
  float p   = (float)pos[s];
  float ang = p * __powf(10000.0f, -(float)(2 * i) / 64.0f);
  float sn, c;
  __sincosf(ang, &sn, &c);
  unsigned short* b = qk + (size_t)m * Dd + dp * 2;
  float x1 = bf2f(b[0]), x2 = bf2f(b[1]);
  b[0] = f2bf_u(x1 * c - x2 * sn);
  b[1] = f2bf_u(x1 * sn + x2 * c);
}

// ---------------------------------------------------------------------------
// 4) Flash attention. Block = 4 waves = 64 query rows of one (b,h).
//    Keys streamed in 32-wide tiles through LDS (async DMA); online softmax.
// ---------------------------------------------------------------------------
__global__ void attn_kernel(const unsigned short* __restrict__ qkv,
                            unsigned short* __restrict__ attn_out) {
  __shared__ __attribute__((aligned(16))) unsigned short kbuf[32][72];
  __shared__ __attribute__((aligned(16))) unsigned short vbufT[64][40];
  __shared__ __attribute__((aligned(16))) unsigned short pbuf[4][16][40];

  const int q0  = blockIdx.x * 64;
  const int h   = blockIdx.y;
  const int b   = blockIdx.z;
  const int tid = threadIdx.x;
  const int w    = tid >> 5;
  const int lane = tid & 31;
  const int lr   = lane & 15;
  const int hi   = lane >> 4;

  const unsigned short* qp = qkv;
  const unsigned short* kp = qkv + (size_t)MB * Dd;
  const unsigned short* vp = qkv + (size_t)2 * MB * Dd;
  const int hoff = h * DH;
  const int qrow = b * Ss + q0 + w * 16;

  // Q fragments (dh split into two K=32 chunks), loaded once.
  FragBF qa[2];
#pragma unroll
  for (int c = 0; c < 2; ++c) {
    const unsigned short* p = qp + (size_t)(qrow + lr) * Dd + hoff + c * 32 + hi * 8;
    qa[c].u[0] = *(const U4*)p;
    qa[c].u[1] = *(const U4*)(p + 16);
  }

  const v8f z8 = {0.f,0.f,0.f,0.f,0.f,0.f,0.f,0.f};
  v8f oacc[4];
#pragma unroll
  for (int t = 0; t < 4; ++t) oacc[t] = z8;
  float mrow[8], lrow[8];
#pragma unroll
  for (int j = 0; j < 8; ++j) { mrow[j] = NEG_BIG; lrow[j] = 0.f; }

  const int nt = (q0 + 64) >> 5;   // key tiles needed (causal)
  for (int t = 0; t < nt; ++t) {
    const int k0 = t * 32;
    __syncthreads();               // previous tile fully consumed

    // K tile: 32 rows x 64 cols -> kbuf via async global->LDS DMA.
#pragma unroll
    for (int r = 0; r < 2; ++r) {
      int id   = tid + r * 128;    // 0..255 b128 chunks
      int row  = id >> 3;
      int col8 = (id & 7) * 8;
      const unsigned short* src = kp + (size_t)(b * Ss + k0 + row) * Dd + hoff + col8;
      async_copy_b128(&kbuf[row][col8], src);
    }
    // V tile transposed: vbufT[dh][key] (element shuffle -> explicit path)
#pragma unroll
    for (int r = 0; r < 8; ++r) {
      int id  = tid * 8 + r;       // 0..1023 bf16-pairs
      int row = id >> 5;           // key 0..31
      int dp  = id & 31;           // dh pair 0..31
      const unsigned short* src = vp + (size_t)(b * Ss + k0 + row) * Dd + hoff + dp * 2;
      unsigned u = *(const unsigned*)src;
      vbufT[dp * 2][row]     = (unsigned short)(u & 0xFFFFu);
      vbufT[dp * 2 + 1][row] = (unsigned short)(u >> 16);
    }
    wait_async0();                 // K DMA landed in LDS
    __syncthreads();

    // Scores: S = Q * K^T, 16x32 per wave (two N-halves).
    v8f s0 = z8, s1 = z8;
#pragma unroll
    for (int c = 0; c < 2; ++c) {
      FragBF kb0, kb1;
      const unsigned short* p0 = &kbuf[lr][c * 32 + hi * 8];
      kb0.u[0] = *(const U4*)p0; kb0.u[1] = *(const U4*)(p0 + 16);
      const unsigned short* p1 = &kbuf[16 + lr][c * 32 + hi * 8];
      kb1.u[0] = *(const U4*)p1; kb1.u[1] = *(const U4*)(p1 + 16);
      s0 = wmma_bf16(qa[c], kb0, s0);
      s1 = wmma_bf16(qa[c], kb1, s1);
    }

    // Online softmax (rows j+8*hi live across the half-wave's 16 lanes).
    float corr[8];
#pragma unroll
    for (int j = 0; j < 8; ++j) {
      int row = q0 + w * 16 + j + 8 * hi;
      float sc0 = (row >= k0 + lr)      ? s0[j] * 0.125f : NEG_BIG;
      float sc1 = (row >= k0 + 16 + lr) ? s1[j] * 0.125f : NEG_BIG;
      float mx = fmaxf(sc0, sc1);
#pragma unroll
      for (int off = 8; off >= 1; off >>= 1)
        mx = fmaxf(mx, __shfl_xor(mx, off, 16));
      float mn = fmaxf(mrow[j], mx);
      corr[j]  = __expf(mrow[j] - mn);
      mrow[j]  = mn;
      float p0 = __expf(sc0 - mn);
      float p1 = __expf(sc1 - mn);
      float rs = p0 + p1;
#pragma unroll
      for (int off = 8; off >= 1; off >>= 1)
        rs += __shfl_xor(rs, off, 16);
      lrow[j] = lrow[j] * corr[j] + rs;
      pbuf[w][j + 8 * hi][lr]      = f2bf_u(p0);
      pbuf[w][j + 8 * hi][16 + lr] = f2bf_u(p1);
    }
#pragma unroll
    for (int tt = 0; tt < 4; ++tt)
#pragma unroll
      for (int j = 0; j < 8; ++j) oacc[tt][j] *= corr[j];

    // P (16x32) as A-fragment from this wave's LDS slice (in-order DS ops).
    FragBF pa;
    pa.u[0] = *(const U4*)&pbuf[w][lr][hi * 8];
    pa.u[1] = *(const U4*)&pbuf[w][lr][16 + hi * 8];
    // PV: 4 output N-tiles over dh.
#pragma unroll
    for (int tt = 0; tt < 4; ++tt) {
      FragBF vb;
      const unsigned short* vq = &vbufT[tt * 16 + lr][hi * 8];
      vb.u[0] = *(const U4*)vq; vb.u[1] = *(const U4*)(vq + 16);
      oacc[tt] = wmma_bf16(pa, vb, oacc[tt]);
    }
  }

  // Normalize and store bf16 attention output [MB][D].
#pragma unroll
  for (int tt = 0; tt < 4; ++tt)
#pragma unroll
    for (int j = 0; j < 8; ++j) {
      float val = oacc[tt][j] / lrow[j];
      int row = qrow + j + 8 * hi;
      int col = hoff + tt * 16 + lr;
      attn_out[(size_t)row * Dd + col] = f2bf_u(val);
    }
}

// ---------------------------------------------------------------------------
// 5) Output projection: out[m][n] = sum_k A[m][k] * Wo[n][k], fp32 store.
//    Same 32x64 wave tile as gemm_qkv.
// ---------------------------------------------------------------------------
__global__ void gemm_proj(const unsigned short* __restrict__ ab,
                          const unsigned short* __restrict__ wob,
                          float* __restrict__ out) {
  const int w    = threadIdx.x >> 5;
  const int lane = threadIdx.x & 31;
  const int lr   = lane & 15;
  const int hi   = lane >> 4;
  const int m0 = (blockIdx.x * 4 + w) * 32;
  const int n0 = blockIdx.y * 64;

  const v8f z8 = {0.f,0.f,0.f,0.f,0.f,0.f,0.f,0.f};
  v8f acc[2][4];
#pragma unroll
  for (int r = 0; r < 2; ++r)
#pragma unroll
    for (int t = 0; t < 4; ++t) acc[r][t] = z8;

  for (int k = 0; k < Dd; k += 32) {
    FragBF a[2];
#pragma unroll
    for (int r = 0; r < 2; ++r) {
      const unsigned short* ap = ab + (size_t)(m0 + r * 16 + lr) * Dd + k + hi * 8;
      a[r].u[0] = *(const U4*)ap;
      a[r].u[1] = *(const U4*)(ap + 16);
    }
#pragma unroll
    for (int t = 0; t < 4; ++t) {
      FragBF bf;
      const unsigned short* bp = wob + (size_t)(n0 + t * 16 + lr) * Dd + k + hi * 8;
      bf.u[0] = *(const U4*)bp;
      bf.u[1] = *(const U4*)(bp + 16);
      acc[0][t] = wmma_bf16(a[0], bf, acc[0][t]);
      acc[1][t] = wmma_bf16(a[1], bf, acc[1][t]);
    }
  }
#pragma unroll
  for (int r = 0; r < 2; ++r)
#pragma unroll
    for (int t = 0; t < 4; ++t)
#pragma unroll
      for (int j = 0; j < 8; ++j) {
        int row = m0 + r * 16 + j + 8 * hi;
        int col = n0 + t * 16 + lr;
        out[(size_t)row * Dd + col] = acc[r][t][j];
      }
}

// ---------------------------------------------------------------------------
extern "C" void kernel_launch(void* const* d_in, const int* in_sizes, int n_in,
                              void* d_out, int out_size, void* d_ws, size_t ws_size,
                              hipStream_t stream) {
  (void)in_sizes; (void)n_in; (void)out_size; (void)ws_size;
  const float* x    = (const float*)d_in[0];   // [4][2048][1024]
  const float* Wqkv = (const float*)d_in[1];   // [3][1024][1024]
  const float* Wo   = (const float*)d_in[2];   // [1024][1024]
  const int*   pos  = (const int*)d_in[3];     // [2048]
  float* out = (float*)d_out;                  // [4][2048][1024] fp32

  const int NX  = MB * Dd;          // 8388608
  const int NWQ = 3 * Dd * Dd;      // 3145728
  const int NWO = Dd * Dd;          // 1048576

  unsigned short* xb    = (unsigned short*)d_ws;
  unsigned short* wqkvb = xb + NX;
  unsigned short* wob   = wqkvb + NWQ;
  unsigned short* qkvb  = wob + NWO;           // 3 * NX
  unsigned short* attnb = qkvb + (size_t)3 * NX;

  convert_f32_bf16<<<NX  / 256, 256, 0, stream>>>(x,    xb,    NX);
  convert_f32_bf16<<<NWQ / 256, 256, 0, stream>>>(Wqkv, wqkvb, NWQ);
  convert_f32_bf16<<<NWO / 256, 256, 0, stream>>>(Wo,   wob,   NWO);

  gemm_qkv<<<dim3(MB / 128, (3 * Dd) / 64), 128, 0, stream>>>(xb, wqkvb, qkvb);

  rope_kernel<<<(2 * MB * (Dd / 2)) / 256, 256, 0, stream>>>(qkvb, pos);

  attn_kernel<<<dim3(Ss / 64, Hh, Bb), 128, 0, stream>>>(qkvb, attnb);

  gemm_proj<<<dim3(MB / 128, Dd / 64), 128, 0, stream>>>(attnb, wob, out);
}
